// MultiHeadAttention_60447369724190
// MI455X (gfx1250) — matrix operand order repaired and checked by reading, not compiled, not run on hardware
//
#include <hip/hip_runtime.h>

// ---------------------------------------------------------------------------
// Problem constants (match the reference)
// ---------------------------------------------------------------------------
#define Bc   4
#define Tc   2048
#define Cc   2048
#define Hc   16
#define HKVc 8
#define Dc   128
#define QKVN 4096            // H*D + 2*HKV*D
#define EPSc 1e-6f
#define SCALEc 0.08838834764831845f   // 1/sqrt(128)

typedef __attribute__((ext_vector_type(16))) __bf16 v16bf;
typedef __attribute__((ext_vector_type(8)))  float  v8f;

union BFrag { v16bf v; unsigned short us[16]; unsigned u[8]; uint4 q[2]; };
union FAcc  { v8f v; float f[8]; };

__device__ __forceinline__ unsigned short f2bf(float f) {
  unsigned u = __float_as_uint(f);
  u += 0x7FFFu + ((u >> 16) & 1u);        // round-to-nearest-even
  return (unsigned short)(u >> 16);
}

// gfx1250 async copy: global -> LDS, 16 bytes per lane, tracked by ASYNCcnt
__device__ __forceinline__ void async_b128(const void* gptr, void* lptr) {
  unsigned lds = (unsigned)(unsigned long long)(uintptr_t)lptr;  // low 32 bits = LDS offset
  unsigned long long ga = (unsigned long long)(uintptr_t)gptr;
  asm volatile("global_load_async_to_lds_b128 %0, %1, off"
               :: "v"(lds), "v"(ga) : "memory");
}
__device__ __forceinline__ void wait_async0() {
  asm volatile("s_wait_asynccnt 0x0" ::: "memory");
}

// ---------------------------------------------------------------------------
// Elementwise fp32 -> bf16
// ---------------------------------------------------------------------------
__global__ void cvt_bf16_kernel(const float* __restrict__ src,
                                unsigned short* __restrict__ dst, long n) {
  long i = (long)blockIdx.x * blockDim.x + threadIdx.x;
  long stride = (long)gridDim.x * blockDim.x;
  for (; i < n; i += stride) dst[i] = f2bf(src[i]);
}

// Pack Wq|Wk|Wv TRANSPOSED into WcT[4096][2048] bf16 (row = out-col n, col = k)
__global__ void pack_wqkv_t_kernel(const float* __restrict__ Wq,
                                   const float* __restrict__ Wk,
                                   const float* __restrict__ Wv,
                                   unsigned short* __restrict__ WcT) {
  long n = (long)QKVN * Cc;
  long i = (long)blockIdx.x * blockDim.x + threadIdx.x;
  long stride = (long)gridDim.x * blockDim.x;
  for (; i < n; i += stride) {
    int row = (int)(i >> 11);        // n: 0..4095
    int k   = (int)(i & 2047);       // k: 0..2047
    float v;
    if (row < 2048)      v = Wq[(size_t)k * 2048 + row];
    else if (row < 3072) v = Wk[(size_t)k * 1024 + (row - 2048)];
    else                 v = Wv[(size_t)k * 1024 + (row - 3072)];
    WcT[i] = f2bf(v);
  }
}

// WoT[n][k] = Wo[k][n], 2048x2048
__global__ void transpose_wo_kernel(const float* __restrict__ Wo,
                                    unsigned short* __restrict__ WoT) {
  long n = (long)Cc * Cc;
  long i = (long)blockIdx.x * blockDim.x + threadIdx.x;
  long stride = (long)gridDim.x * blockDim.x;
  for (; i < n; i += stride) {
    int row = (int)(i >> 11);        // n
    int k   = (int)(i & 2047);       // k
    WoT[i] = f2bf(Wo[(size_t)k * Cc + row]);
  }
}

__global__ void pack_bias_kernel(const float* __restrict__ bq,
                                 const float* __restrict__ bk,
                                 const float* __restrict__ bv,
                                 float* __restrict__ bc) {
  int i = blockIdx.x * blockDim.x + threadIdx.x;
  if (i >= QKVN) return;
  float v;
  if (i < 2048)      v = bq[i];
  else if (i < 3072) v = bk[i - 2048];
  else               v = bv[i - 3072];
  bc[i] = v;
}

__global__ void lengths_kernel(const int* __restrict__ mask, int* __restrict__ len) {
  __shared__ int sm[256];
  int b = blockIdx.x;
  int s = 0;
  for (int i = threadIdx.x; i < Tc; i += blockDim.x) s += mask[b * Tc + i];
  sm[threadIdx.x] = s;
  __syncthreads();
  for (int o = 128; o > 0; o >>= 1) {
    if (threadIdx.x < o) sm[threadIdx.x] += sm[threadIdx.x + o];
    __syncthreads();
  }
  if (threadIdx.x == 0) len[b] = sm[0];
}

// ---------------------------------------------------------------------------
// bf16 WMMA GEMM: C[M,N] (fp32) = A[M,K] * BT[N,K]^T + bias
// 256 threads = 8 waves; block tile 128x128; wave tile 64x32; K-step 32.
// Async-copy staged, ping-pong LDS double buffer.
// ---------------------------------------------------------------------------
__global__ void __launch_bounds__(256)
gemm_bf16_kernel(const unsigned short* __restrict__ A,   // [M][K]
                 const unsigned short* __restrict__ BT,  // [N][K]
                 const float* __restrict__ bias,
                 float* __restrict__ C,
                 int M, int N, int K) {
  __shared__ unsigned short As[2][128 * 40];   // [m][k], padded rows
  __shared__ unsigned short Bs[2][128 * 40];   // [n][k], padded rows

  const int tid  = threadIdx.x;
  const int w    = tid >> 5;
  const int lane = tid & 31;
  const int m0   = blockIdx.y * 128;
  const int n0   = blockIdx.x * 128;
  const int waveM = (w & 1) * 64;
  const int waveN = (w >> 1) * 32;
  const int hi = lane >> 4;
  const int l16 = lane & 15;

  const unsigned short* Ab = A  + (size_t)m0 * K;
  const unsigned short* Bb = BT + (size_t)n0 * K;

  // stage one 128x32 tile (512 x 16B chunks; 2 per thread) via async copy
  auto stage = [&](unsigned short* sdst, const unsigned short* gsrc) {
#pragma unroll
    for (int c = tid; c < 512; c += 256) {
      int row = c >> 2, part = c & 3;          // 4 chunks of 8 bf16 per row
      async_b128(&gsrc[(size_t)row * K + part * 8], &sdst[row * 40 + part * 8]);
    }
  };

  FAcc acc[4][2];
#pragma unroll
  for (int mt = 0; mt < 4; ++mt)
#pragma unroll
    for (int nt = 0; nt < 2; ++nt)
#pragma unroll
      for (int r = 0; r < 8; ++r) acc[mt][nt].f[r] = 0.0f;

  stage(As[0], Ab);
  stage(Bs[0], Bb);
  wait_async0();
  __syncthreads();

  int cur = 0;
  for (int k0 = 0; k0 < K; k0 += 32) {
    const int nxt = cur ^ 1;
    const bool more = (k0 + 32 < K);
    if (more) {                                 // prefetch next tile (async)
      stage(As[nxt], Ab + k0 + 32);
      stage(Bs[nxt], Bb + k0 + 32);
    }

    BFrag af[4];
#pragma unroll
    for (int mt = 0; mt < 4; ++mt) {
      const unsigned short* p = &As[cur][(waveM + mt * 16 + l16) * 40 + hi * 8];
      af[mt].q[0] = *(const uint4*)p;           // K pairs 0..7   (this half)
      af[mt].q[1] = *(const uint4*)(p + 16);    // K pairs 16..23 (this half)
    }
    BFrag bfr[2];
#pragma unroll
    for (int nt = 0; nt < 2; ++nt) {
      const unsigned short* p = &Bs[cur][(waveN + nt * 16 + l16) * 40 + hi * 8];
      bfr[nt].q[0] = *(const uint4*)p;          // K = hi*8 + 0..7
      bfr[nt].q[1] = *(const uint4*)(p + 16);   // K = 16 + hi*8 + 0..7
    }
#pragma unroll
    for (int mt = 0; mt < 4; ++mt)
#pragma unroll
      for (int nt = 0; nt < 2; ++nt)
        acc[mt][nt].v = __builtin_amdgcn_wmma_f32_16x16x32_bf16(
            false, af[mt].v, false, bfr[nt].v, (short)0, acc[mt][nt].v, false, false);

    if (more) wait_async0();
    __syncthreads();
    cur = nxt;
  }

  // epilogue: D layout (m = r + 8*hi, n = lane&15)
#pragma unroll
  for (int mt = 0; mt < 4; ++mt)
#pragma unroll
    for (int nt = 0; nt < 2; ++nt)
#pragma unroll
      for (int r = 0; r < 8; ++r) {
        int row = m0 + waveM + mt * 16 + r + 8 * hi;
        int col = n0 + waveN + nt * 16 + l16;
        float v = acc[mt][nt].f[r];
        if (bias) v += bias[col];
        C[(size_t)row * N + col] = v;
      }
}

// ---------------------------------------------------------------------------
// RMSNorm + mRoPE (+mask): qkv fp32 -> q/k bf16 [B,head,T,D], v bf16 [B,hkv,D,T]
// ---------------------------------------------------------------------------
__global__ void rmsrope_kernel(const float* __restrict__ qkv,
                               const float* __restrict__ cosb,
                               const float* __restrict__ sinb,
                               const int* __restrict__ mask,
                               const float* __restrict__ gq,
                               const float* __restrict__ gk,
                               unsigned short* __restrict__ qbf,
                               unsigned short* __restrict__ kbf,
                               unsigned short* __restrict__ vbfT) {
  int bt = blockIdx.x;
  int b = bt >> 11;          // /T
  int t = bt & (Tc - 1);
  int tid = threadIdx.x, wave = tid >> 5, lane = tid & 31;
  size_t base = (size_t)bt * QKVN;
  float mk = (float)mask[bt];

  // mRoPE section reorder (SECTIONS 24/20/20, duplicated)
  int dm0 = lane;               int ax0 = (dm0 < 24) ? 0 : 1;
  int dm1 = lane + 32;          int ax1 = (dm1 < 44) ? 1 : 2;
  size_t cI0 = ((size_t)(ax0 * Bc + b) * Tc + t) * 64 + dm0;
  size_t cI1 = ((size_t)(ax1 * Bc + b) * Tc + t) * 64 + dm1;
  float c0 = cosb[cI0], s0 = sinb[cI0];
  float c1 = cosb[cI1], s1 = sinb[cI1];

  for (int it = 0; it < 4; ++it) {
    int row = wave + 8 * it;                       // 0..31
    float v0, v1, v2, v3, o0, o1, o2, o3;
    if (row < 24) {  // q or k: RMSNorm + RoPE
      int off; const float* g; float mm; unsigned short* dst;
      if (row < 16) {
        off = row * Dc; g = gq; mm = 1.0f;
        dst = qbf + ((size_t)(b * Hc + row) * Tc + t) * Dc;
      } else {
        int h = row - 16; off = 2048 + h * Dc; g = gk; mm = mk;
        dst = kbf + ((size_t)(b * HKVc + h) * Tc + t) * Dc;
      }
      v0 = qkv[base + off + lane];
      v1 = qkv[base + off + lane + 32];
      v2 = qkv[base + off + lane + 64];
      v3 = qkv[base + off + lane + 96];
      float ss = v0 * v0 + v1 * v1 + v2 * v2 + v3 * v3;
#pragma unroll
      for (int o = 1; o < 32; o <<= 1) ss += __shfl_xor(ss, o, 32);
      float rr = rsqrtf(ss * (1.0f / 128.0f) + EPSc);
      float y0 = g[lane]      * v0 * rr;
      float y1 = g[lane + 32] * v1 * rr;
      float y2 = g[lane + 64] * v2 * rr;
      float y3 = g[lane + 96] * v3 * rr;
      o0 = y0 * c0 - y2 * s0;
      o1 = y1 * c1 - y3 * s1;
      o2 = y2 * c0 + y0 * s0;
      o3 = y3 * c1 + y1 * s1;
      dst[lane]      = f2bf(o0 * mm);
      dst[lane + 32] = f2bf(o1 * mm);
      dst[lane + 64] = f2bf(o2 * mm);
      dst[lane + 96] = f2bf(o3 * mm);
    } else {         // v: mask only, stored TRANSPOSED [D][T]
      int h = row - 24; int off = 3072 + h * Dc;
      unsigned short* dst = vbfT + ((size_t)(b * HKVc + h) * Dc) * Tc + t;
      v0 = qkv[base + off + lane];
      v1 = qkv[base + off + lane + 32];
      v2 = qkv[base + off + lane + 64];
      v3 = qkv[base + off + lane + 96];
      dst[(size_t)(lane)      * Tc] = f2bf(v0 * mk);
      dst[(size_t)(lane + 32) * Tc] = f2bf(v1 * mk);
      dst[(size_t)(lane + 64) * Tc] = f2bf(v2 * mk);
      dst[(size_t)(lane + 96) * Tc] = f2bf(v3 * mk);
    }
  }
}

// ---------------------------------------------------------------------------
// Flash attention (causal, GQA repeats=2) with WMMA bf16 + async staging.
// Block = (b, h, 128-row q tile); 8 waves x 16 q rows each.
// ---------------------------------------------------------------------------
__global__ void __launch_bounds__(256)
flash_attn_kernel(const unsigned short* __restrict__ qbf,
                  const unsigned short* __restrict__ kbf,   // [B][HKV][T][D]
                  const unsigned short* __restrict__ vbfT,  // [B][HKV][D][T]
                  const int* __restrict__ lengths,
                  unsigned short* __restrict__ att) {
  __shared__ unsigned short Ks[32 * 136];   // [key][d], padded
  __shared__ unsigned short Vt[128 * 40];   // [d][key], padded
  __shared__ unsigned short Ps[8 * 512];    // per-wave 16x32 prob staging

  const int b  = blockIdx.z;
  const int h  = blockIdx.y;
  const int hkv = h >> 1;                   // repeats = H/HKV = 2
  const int q0 = blockIdx.x * 128;
  const int tid = threadIdx.x, w = tid >> 5, lane = tid & 31;
  const int hi = lane >> 4, l16 = lane & 15;
  const int qbase = q0 + w * 16;
  const int len = lengths[b];

  const unsigned short* kb = kbf  + (size_t)(b * HKVc + hkv) * Tc * Dc;
  const unsigned short* vb = vbfT + (size_t)(b * HKVc + hkv) * Dc * Tc;

  // persistent Q A-fragments (b128 halves are contiguous)
  BFrag aq[4];
  {
    const unsigned short* rp = qbf + ((size_t)(b * Hc + h) * Tc + (qbase + l16)) * Dc;
#pragma unroll
    for (int dt = 0; dt < 4; ++dt) {
      aq[dt].q[0] = *(const uint4*)&rp[dt * 32 + hi * 8];
      aq[dt].q[1] = *(const uint4*)&rp[dt * 32 + 16 + hi * 8];
    }
  }

  FAcc acc[8];
  float mstate[8], lstate[8];
#pragma unroll
  for (int i = 0; i < 8; ++i) {
    mstate[i] = -3e38f; lstate[i] = 0.0f;
#pragma unroll
    for (int r = 0; r < 8; ++r) acc[i].f[r] = 0.0f;
  }

  unsigned short* Pw = &Ps[w * 512];
  const int kend = q0 + 128;                       // causal bound for this block

  for (int kt = 0; kt < kend; kt += 32) {
    // stage K tile [32 keys][128 d] via async copy (2 x b128 per thread)
#pragma unroll
    for (int c = tid; c < 512; c += 256) {
      int key = c >> 4, part = c & 15;
      async_b128(&kb[(size_t)(kt + key) * Dc + part * 8], &Ks[key * 136 + part * 8]);
    }
    // stage V tile [128 d][32 keys] (V already transposed in memory)
#pragma unroll
    for (int c = tid; c < 512; c += 256) {
      int d = c >> 2, part = c & 3;
      async_b128(&vb[(size_t)d * Tc + kt + part * 8], &Vt[d * 40 + part * 8]);
    }
    wait_async0();
    __syncthreads();

    // scores: S[16x32] = Q[16x128] * K^T
    FAcc s0, s1;
#pragma unroll
    for (int r = 0; r < 8; ++r) { s0.f[r] = 0.0f; s1.f[r] = 0.0f; }
#pragma unroll
    for (int nt = 0; nt < 2; ++nt) {
      int keyLoc = nt * 16 + l16;
#pragma unroll
      for (int dt = 0; dt < 4; ++dt) {
        BFrag bk;
        const unsigned short* p = &Ks[keyLoc * 136 + dt * 32 + hi * 8];
        bk.q[0] = *(const uint4*)p;
        bk.q[1] = *(const uint4*)(p + 16);
        if (nt == 0)
          s0.v = __builtin_amdgcn_wmma_f32_16x16x32_bf16(false, aq[dt].v, false, bk.v,
                                                         (short)0, s0.v, false, false);
        else
          s1.v = __builtin_amdgcn_wmma_f32_16x16x32_bf16(false, aq[dt].v, false, bk.v,
                                                         (short)0, s1.v, false, false);
      }
    }

    // online softmax per row (rows live in 16-lane halves)
    float ascale[8];
#pragma unroll
    for (int r = 0; r < 8; ++r) {
      int qrow = qbase + r + 8 * hi;
      int key0 = kt + l16, key1 = key0 + 16;
      float v0 = s0.f[r] * SCALEc, v1 = s1.f[r] * SCALEc;
      if (key0 > qrow || key0 >= len) v0 = -1e9f;
      if (key1 > qrow || key1 >= len) v1 = -1e9f;
      float rm = fmaxf(v0, v1);
#pragma unroll
      for (int o = 1; o < 16; o <<= 1) rm = fmaxf(rm, __shfl_xor(rm, o, 32));
      float nm = fmaxf(mstate[r], rm);
      float al = __expf(mstate[r] - nm);
      float p0 = __expf(v0 - nm), p1 = __expf(v1 - nm);
      float rs = p0 + p1;
#pragma unroll
      for (int o = 1; o < 16; o <<= 1) rs += __shfl_xor(rs, o, 32);
      lstate[r] = lstate[r] * al + rs;
      mstate[r] = nm;
      ascale[r] = al;
      int m = r + 8 * hi;
      Pw[m * 32 + l16]      = f2bf(p0);
      Pw[m * 32 + 16 + l16] = f2bf(p1);
    }
#pragma unroll
    for (int dt2 = 0; dt2 < 8; ++dt2)
#pragma unroll
      for (int r = 0; r < 8; ++r) acc[dt2].f[r] *= ascale[r];

    asm volatile("s_wait_dscnt 0x0" ::: "memory");   // same-wave LDS store->load

    // reload P in A-fragment layout
    BFrag pa;
    {
      const unsigned short* p = &Pw[l16 * 32 + hi * 8];
      pa.q[0] = *(const uint4*)p;
      pa.q[1] = *(const uint4*)(p + 16);
    }
    // out += P[16x32] * V[32x128]
#pragma unroll
    for (int dt2 = 0; dt2 < 8; ++dt2) {
      BFrag bvf;
      const unsigned short* p = &Vt[(dt2 * 16 + l16) * 40 + hi * 8];
      bvf.q[0] = *(const uint4*)p;
      bvf.q[1] = *(const uint4*)(p + 16);
      acc[dt2].v = __builtin_amdgcn_wmma_f32_16x16x32_bf16(false, pa.v, false, bvf.v,
                                                           (short)0, acc[dt2].v, false, false);
    }
    __syncthreads();
  }

  // epilogue: att[b, qrow, h*128 + d] bf16
#pragma unroll
  for (int r = 0; r < 8; ++r) {
    int qrow = qbase + r + 8 * hi;
    float inv = 1.0f / lstate[r];
#pragma unroll
    for (int dt2 = 0; dt2 < 8; ++dt2) {
      float v = acc[dt2].f[r] * inv;
      att[((size_t)(b * Tc + qrow)) * (Hc * Dc) + h * Dc + dt2 * 16 + l16] = f2bf(v);
    }
  }
}

// ---------------------------------------------------------------------------
// Host-side orchestration
// ---------------------------------------------------------------------------
extern "C" void kernel_launch(void* const* d_in, const int* in_sizes, int n_in,
                              void* d_out, int out_size, void* d_ws, size_t ws_size,
                              hipStream_t stream) {
  const float* x    = (const float*)d_in[0];
  const float* cosb = (const float*)d_in[1];
  const float* sinb = (const float*)d_in[2];
  const int*   mask = (const int*)  d_in[3];
  const float* Wq   = (const float*)d_in[4];
  const float* bq   = (const float*)d_in[5];
  const float* Wk   = (const float*)d_in[6];
  const float* bk   = (const float*)d_in[7];
  const float* Wv   = (const float*)d_in[8];
  const float* bv   = (const float*)d_in[9];
  const float* gq   = (const float*)d_in[10];
  const float* gk   = (const float*)d_in[11];
  const float* Wo   = (const float*)d_in[12];

  const size_t M = (size_t)Bc * Tc;           // 8192
  char* ws = (char*)d_ws;
  size_t off = 0;
  auto carve = [&](size_t bytes) { size_t r = off; off += (bytes + 255) & ~(size_t)255; return r; };

  unsigned short* xbf  = (unsigned short*)(ws + carve(M * Cc * 2));
  unsigned short* wcbf = (unsigned short*)(ws + carve((size_t)Cc * QKVN * 2));   // WcT [4096][2048]
  unsigned short* wobf = (unsigned short*)(ws + carve((size_t)Cc * Cc * 2));     // WoT [2048][2048]
  float*          bc   = (float*)         (ws + carve(QKVN * 4));
  size_t qkvOff = carve(M * QKVN * 4);
  float*          qkv  = (float*)(ws + qkvOff);
  unsigned short* attbf= (unsigned short*)(ws + qkvOff);                          // alias (qkv dead by then)
  unsigned short* qbf  = (unsigned short*)(ws + carve((size_t)Bc * Hc   * Tc * Dc * 2));
  unsigned short* kbf  = (unsigned short*)(ws + carve((size_t)Bc * HKVc * Tc * Dc * 2));
  unsigned short* vbfT = (unsigned short*)(ws + carve((size_t)Bc * HKVc * Tc * Dc * 2));
  int*            len  = (int*)(ws + carve(256));
  (void)ws_size; (void)in_sizes; (void)n_in; (void)out_size;

  // 1) conversions / packing (weights pre-transposed for K-contiguous B tiles)
  cvt_bf16_kernel   <<<4096, 256, 0, stream>>>(x, xbf, (long)M * Cc);
  pack_wqkv_t_kernel<<<4096, 256, 0, stream>>>(Wq, Wk, Wv, wcbf);
  transpose_wo_kernel<<<2048, 256, 0, stream>>>(Wo, wobf);
  pack_bias_kernel  <<<16, 256, 0, stream>>>(bq, bk, bv, bc);
  lengths_kernel    <<<Bc, 256, 0, stream>>>(mask, len);

  // 2) fused QKV projection: [8192 x 4096] = x[8192 x 2048] * WcT^T
  gemm_bf16_kernel<<<dim3(QKVN / 128, M / 128), 256, 0, stream>>>(
      xbf, wcbf, bc, qkv, (int)M, QKVN, Cc);

  // 3) RMSNorm + RoPE + mask -> bf16 q/k (row-major) and v (transposed)
  rmsrope_kernel<<<(int)M, 256, 0, stream>>>(qkv, cosb, sinb, mask, gq, gk, qbf, kbf, vbfT);

  // 4) causal flash attention (GQA) -> att bf16 [8192 x 2048]
  flash_attn_kernel<<<dim3(Tc / 128, Hc, Bc), 256, 0, stream>>>(qbf, kbf, vbfT, len, attbf);

  // 5) output projection -> fp32 d_out
  gemm_bf16_kernel<<<dim3(Cc / 128, M / 128), 256, 0, stream>>>(
      attbf, wobf, nullptr, (float*)d_out, (int)M, Cc, Cc);
}
